// TFMultiHeadAttention_52329881535060
// MI455X (gfx1250) — matrix operand order repaired and checked by reading, not compiled, not run on hardware
//
#include <hip/hip_runtime.h>
#include <hip/hip_bf16.h>
#include <stdint.h>

// ---------------------------------------------------------------------------
// CTRL-style multi-head attention forward, bf16 WMMA (gfx1250 / MI455X).
// B=2, S=2048, D=1024, H=16, DH=64.
// Workspace (bf16 elements), ~40 MB:
//   WqT/WkT/WvT/WoT (1Mx4), qh[B,H,S,DH], kh[B,H,S,DH], vT[B,H,DH,S], ctx.
// d_out: out [B,S,D] fp32 then present [B,2,H,S,DH] fp32.
// ---------------------------------------------------------------------------

#define D_MODEL 1024
#define SEQ     2048
#define NB      2
#define NH      16
#define DHEAD   64

typedef __attribute__((ext_vector_type(16))) __bf16 bf16x16;
typedef __attribute__((ext_vector_type(8)))  float  f32x8;

struct __align__(16) U128 { unsigned int x[4]; };

union Frag16 {
  bf16x16        v;
  U128           q[2];
  unsigned short u[16];
  __bf16         h[16];
};

__device__ inline unsigned short f2bf(float f) {
  union { __bf16 h; unsigned short u; } cv;
  cv.h = (__bf16)f;                      // native cvt, RNE
  return cv.u;
}

__device__ inline f32x8 wmma_bf16(const Frag16& a, const Frag16& b, f32x8 c) {
  return __builtin_amdgcn_wmma_f32_16x16x32_bf16(
      /*neg_a=*/false, a.v, /*neg_b=*/false, b.v,
      /*c_mod=*/(short)0, c, /*reuse_a=*/false, /*reuse_b=*/false);
}

// A-matrix 16x32 bf16 fragment: lane<16 holds K {k0..k0+7, k0+16..k0+23},
// lane>=16 holds K {k0+8..k0+15, k0+24..k0+31} of its row.
__device__ inline void load_afrag(Frag16& a, const unsigned short* row, int k0, int half) {
  int ka = k0 + (half ? 8 : 0);
  a.q[0] = *(const U128*)(row + ka);
  a.q[1] = *(const U128*)(row + ka + 16);
}

// B-matrix 32x16 bf16 fragment: lane<16 holds K {k0..k0+7, k0+16..k0+23} of column (lane&15),
// lane>=16 holds K {k0+8..k0+15, k0+24..k0+31}.
__device__ inline void load_bfrag(Frag16& b, const unsigned short* col, int k0, int half) {
  int kb = k0 + (half ? 8 : 0);
  b.q[0] = *(const U128*)(col + kb);
  b.q[1] = *(const U128*)(col + kb + 16);
}

// Async copy: 512B per instruction (32 lanes x 16B), GVS addressing.
// VDST = LDS byte offset, VADDR = per-lane byte offset, SADDR = 64-bit base.
__device__ inline void async_copy_b128(unsigned lds_off, unsigned gbl_off,
                                       const void* base) {
  asm volatile("global_load_async_to_lds_b128 %0, %1, %2"
               :: "v"(lds_off), "v"(gbl_off), "s"(base) : "memory");
}
__device__ inline void async_wait0() {
  asm volatile("s_wait_asynccnt 0" ::: "memory");
}

// ---------------------------------------------------------------------------
// Kernel 1: W (fp32 row-major [D,D]) -> WT (bf16 [N rows][K]) transpose.
// ---------------------------------------------------------------------------
__global__ void transpose_w_bf16(const float* __restrict__ W,
                                 unsigned short* __restrict__ WT) {
  int idx = blockIdx.x * blockDim.x + threadIdx.x;
  int r = idx >> 10;
  int c = idx & 1023;
  WT[(size_t)c * D_MODEL + r] = f2bf(W[idx]);
}

// ---------------------------------------------------------------------------
// Kernel 2: Y = X @ W + b. One 16x16 tile per wave, K step 32, bf16 WMMA.
// vmode 0: Yh[b,h,s,dh]; vmode 1: Yh[b,h,dh,s]. present (fp32) optional.
// ---------------------------------------------------------------------------
__global__ void qkv_proj(const float* __restrict__ X,
                         const unsigned short* __restrict__ WT,
                         const float* __restrict__ bias,
                         unsigned short* __restrict__ Yh,
                         float* __restrict__ present,
                         int vmode) {
  int wave = blockIdx.x * blockDim.y + threadIdx.y;   // 0..16383
  int tm = wave >> 6;
  int tn = wave & 63;
  int lane = threadIdx.x;
  int half = lane >> 4;
  int lsub = lane & 15;

  int arow = tm * 16 + lsub;
  int bcol = tn * 16 + lsub;
  const float* xrow = X + (size_t)arow * D_MODEL;
  const unsigned short* wcol = WT + (size_t)bcol * D_MODEL;

  f32x8 acc = {};
  for (int k0 = 0; k0 < D_MODEL; k0 += 32) {
    if (k0 + 32 < D_MODEL) {                 // global_prefetch_b8 next K block
      __builtin_prefetch(xrow + k0 + 32, 0, 3);
      __builtin_prefetch(wcol + k0 + 32, 0, 3);
    }
    Frag16 a, b;
    int ka = k0 + (half ? 8 : 0);
    float4 f0 = *(const float4*)(xrow + ka);
    float4 f1 = *(const float4*)(xrow + ka + 4);
    float4 f2 = *(const float4*)(xrow + ka + 16);
    float4 f3 = *(const float4*)(xrow + ka + 20);
    a.h[0]  = (__bf16)f0.x; a.h[1]  = (__bf16)f0.y; a.h[2]  = (__bf16)f0.z; a.h[3]  = (__bf16)f0.w;
    a.h[4]  = (__bf16)f1.x; a.h[5]  = (__bf16)f1.y; a.h[6]  = (__bf16)f1.z; a.h[7]  = (__bf16)f1.w;
    a.h[8]  = (__bf16)f2.x; a.h[9]  = (__bf16)f2.y; a.h[10] = (__bf16)f2.z; a.h[11] = (__bf16)f2.w;
    a.h[12] = (__bf16)f3.x; a.h[13] = (__bf16)f3.y; a.h[14] = (__bf16)f3.z; a.h[15] = (__bf16)f3.w;
    load_bfrag(b, wcol, k0, half);
    acc = wmma_bf16(a, b, acc);
  }

  float bval = bias[bcol];
  int h  = bcol >> 6;
  int dh = bcol & 63;
#pragma unroll
  for (int j = 0; j < 8; ++j) {
    int row = tm * 16 + j + (half ? 8 : 0);
    int bb = row >> 11;
    int s  = row & 2047;
    float val = acc[j] + bval;
    if (vmode == 0) {
      Yh[(((size_t)(bb * NH + h) * SEQ + s) * DHEAD) + dh] = f2bf(val);
    } else {
      Yh[(((size_t)(bb * NH + h) * DHEAD + dh) * SEQ) + s] = f2bf(val);
    }
    if (present) {
      present[((((size_t)(bb * 2 + vmode) * NH + h) * SEQ + s) * DHEAD) + dh] = val;
    }
  }
}

// ---------------------------------------------------------------------------
// Kernel 3: flash attention. Block = 4 waves = 4 consecutive query tiles of
// one (b,h). K/V 32-key tiles staged to LDS via global_load_async_to_lds_b128
// (ASYNCcnt), shared by all 4 waves. Wave-uniform causal skip keeps EXEC all
// ones at every WMMA; barriers are in uniform control flow.
// ---------------------------------------------------------------------------
__global__ void flash_attn(const unsigned short* __restrict__ qh,
                           const unsigned short* __restrict__ kh,
                           const unsigned short* __restrict__ vT,
                           unsigned short* __restrict__ ctx) {
  __shared__ __align__(16) unsigned short kbuf[32 * DHEAD];   // 4KB  [key][dh]
  __shared__ __align__(16) unsigned short vbuf[DHEAD * 32];   // 4KB  [dh][key]
  __shared__ __align__(16) unsigned short plds[4][16 * 32];   // 1KB/wave

  int wid  = threadIdx.y;
  int lane = threadIdx.x;
  int half = lane >> 4;
  int lsub = lane & 15;

  int bh = blockIdx.x >> 5;          // 0..31 (b*H + h)
  int qg = blockIdx.x & 31;          // query group (4 tiles each)
  int qt = qg * 4 + wid;
  int q0 = qt * 16;

  const unsigned short* qbase = qh + (size_t)bh * SEQ * DHEAD;
  const unsigned short* kbase = kh + (size_t)bh * SEQ * DHEAD;
  const unsigned short* vbase = vT + (size_t)bh * DHEAD * SEQ;
  unsigned short* pl = plds[wid];

  unsigned kbuf_lds = (unsigned)(uintptr_t)&kbuf[0];
  unsigned vbuf_lds = (unsigned)(uintptr_t)&vbuf[0];

  // Q fragments for dh 0..31 and 32..63 (loaded once, per wave)
  Frag16 aq[2];
  {
    const unsigned short* qrow = qbase + (size_t)(q0 + lsub) * DHEAD;
    load_afrag(aq[0], qrow, 0, half);
    load_afrag(aq[1], qrow, 32, half);
  }

  f32x8 zero = {};
  f32x8 acc[4];
  acc[0] = zero; acc[1] = zero; acc[2] = zero; acc[3] = zero;
  float mrun[8], lrun[8];
#pragma unroll
  for (int j = 0; j < 8; ++j) { mrun[j] = -3.0e38f; lrun[j] = 0.0f; }

  int nkb_blk = 2 * qg + 2;          // key blocks needed by highest wave
  for (int kb = 0; kb < nkb_blk; ++kb) {
    int s0 = kb * 32;

    __syncthreads();                 // previous iteration's LDS reads done
    // ---- stage K tile (contiguous 4KB) and V tile (64 rows x 64B) ----
#pragma unroll
    for (int i = 0; i < 2; ++i) {
      int c = wid * 2 + i;                         // chunk 0..7, 512B each
      unsigned bo = (unsigned)(c * 512 + lane * 16);
      async_copy_b128(kbuf_lds + bo, (unsigned)(s0 * DHEAD * 2) + bo, kbase);
    }
#pragma unroll
    for (int i = 0; i < 2; ++i) {
      int c = wid * 2 + i;                         // chunk 0..7
      int row = c * 8 + (lane >> 2);               // dh row 0..63
      int piece = lane & 3;                        // 16B piece within 64B row
      async_copy_b128(vbuf_lds + (unsigned)((row * 32 + piece * 8) * 2),
                      (unsigned)((row * SEQ + s0) * 2 + piece * 16), vbase);
    }
    async_wait0();
    __syncthreads();                 // tiles visible to all waves

    if (s0 <= q0 + 15) {             // wave-uniform causal skip
      // ---- logits: 16 queries x 32 keys, two C tiles ----
      f32x8 sc[2];
      sc[0] = zero; sc[1] = zero;
#pragma unroll
      for (int t = 0; t < 2; ++t) {
        const unsigned short* krow = kbuf + (size_t)(t * 16 + lsub) * DHEAD;
        Frag16 bk0, bk1;
        load_bfrag(bk0, krow, 0, half);
        load_bfrag(bk1, krow, 32, half);
        sc[t] = wmma_bf16(aq[0], bk0, sc[t]);
        sc[t] = wmma_bf16(aq[1], bk1, sc[t]);
      }

      // ---- scale, causal mask, online softmax ----
      float p0a[8], p1a[8];
#pragma unroll
      for (int j = 0; j < 8; ++j) {
        int row = q0 + j + (half ? 8 : 0);
        int key0 = s0 + lsub;
        float v0 = sc[0][j] * 0.125f;
        float v1 = sc[1][j] * 0.125f;
        if (key0 > row)      v0 -= 10000.0f;
        if (key0 + 16 > row) v1 -= 10000.0f;
        float rm = fmaxf(v0, v1);
        rm = fmaxf(rm, __shfl_xor(rm, 1));
        rm = fmaxf(rm, __shfl_xor(rm, 2));
        rm = fmaxf(rm, __shfl_xor(rm, 4));
        rm = fmaxf(rm, __shfl_xor(rm, 8));
        float mnew  = fmaxf(mrun[j], rm);
        float alpha = __expf(mrun[j] - mnew);
        float p0 = __expf(v0 - mnew);
        float p1 = __expf(v1 - mnew);
        float rs = p0 + p1;
        rs += __shfl_xor(rs, 1);
        rs += __shfl_xor(rs, 2);
        rs += __shfl_xor(rs, 4);
        rs += __shfl_xor(rs, 8);
        lrun[j] = lrun[j] * alpha + rs;
        mrun[j] = mnew;
        acc[0][j] *= alpha; acc[1][j] *= alpha;
        acc[2][j] *= alpha; acc[3][j] *= alpha;
        p0a[j] = p0; p1a[j] = p1;
      }

      // ---- P: C-layout -> A-layout via per-wave LDS scratch ----
#pragma unroll
      for (int j = 0; j < 8; ++j) {
        int prow = j + (half ? 8 : 0);
        pl[prow * 32 + lsub]      = f2bf(p0a[j]);
        pl[prow * 32 + 16 + lsub] = f2bf(p1a[j]);
      }
      Frag16 pa;
      load_afrag(pa, pl + lsub * 32, 0, half);

      // ---- PV: P(16x32) x V(32x64) as four 16x16 dh-chunks ----
#pragma unroll
      for (int t = 0; t < 4; ++t) {
        const unsigned short* vrow = vbuf + (size_t)(t * 16 + lsub) * 32;
        Frag16 bv;
        load_bfrag(bv, vrow, 0, half);
        acc[t] = wmma_bf16(pa, bv, acc[t]);
      }
    }
  }

  // ---- normalize and write ctx bf16 [b,h,s,dh] ----
  unsigned short* cbase = ctx + (size_t)bh * SEQ * DHEAD;
#pragma unroll
  for (int t = 0; t < 4; ++t) {
    int dh = t * 16 + lsub;
#pragma unroll
    for (int j = 0; j < 8; ++j) {
      int row = q0 + j + (half ? 8 : 0);
      cbase[(size_t)row * DHEAD + dh] = f2bf(acc[t][j] / lrun[j]);
    }
  }
}

// ---------------------------------------------------------------------------
// Kernel 4: out = merge_heads(ctx) @ Wo + bo, fp32 output.
// ---------------------------------------------------------------------------
__global__ void out_proj(const unsigned short* __restrict__ ctx,
                         const unsigned short* __restrict__ WoT,
                         const float* __restrict__ bias,
                         float* __restrict__ out) {
  int wave = blockIdx.x * blockDim.y + threadIdx.y;
  int tm = wave >> 6;
  int tn = wave & 63;
  int lane = threadIdx.x;
  int half = lane >> 4;
  int lsub = lane & 15;

  int arow = tm * 16 + lsub;
  int bb = arow >> 11;
  int s  = arow & 2047;
  int bcol = tn * 16 + lsub;
  const unsigned short* wcol = WoT + (size_t)bcol * D_MODEL;

  f32x8 acc = {};
  for (int k0 = 0; k0 < D_MODEL; k0 += 32) {
    if (k0 + 32 < D_MODEL) __builtin_prefetch(wcol + k0 + 32, 0, 3);
    int h   = k0 >> 6;
    int dh0 = k0 & 63;
    const unsigned short* crow = ctx + ((size_t)(bb * NH + h) * SEQ + s) * DHEAD;
    Frag16 a, b;
    load_afrag(a, crow, dh0, half);
    load_bfrag(b, wcol, k0, half);
    acc = wmma_bf16(a, b, acc);
  }

  float bval = bias[bcol];
#pragma unroll
  for (int j = 0; j < 8; ++j) {
    int row = tm * 16 + j + (half ? 8 : 0);
    out[(size_t)row * D_MODEL + bcol] = acc[j] + bval;
  }
}

// ---------------------------------------------------------------------------
extern "C" void kernel_launch(void* const* d_in, const int* in_sizes, int n_in,
                              void* d_out, int out_size, void* d_ws, size_t ws_size,
                              hipStream_t stream) {
  (void)in_sizes; (void)n_in; (void)out_size; (void)ws_size;

  const float* q  = (const float*)d_in[0];
  const float* k  = (const float*)d_in[1];
  const float* v  = (const float*)d_in[2];
  /* d_in[3] = mask: causal, hardcoded in flash_attn */
  const float* Wq = (const float*)d_in[4];
  const float* bq = (const float*)d_in[5];
  const float* Wk = (const float*)d_in[6];
  const float* bk = (const float*)d_in[7];
  const float* Wv = (const float*)d_in[8];
  const float* bv = (const float*)d_in[9];
  const float* Wo = (const float*)d_in[10];
  const float* bo = (const float*)d_in[11];

  float* out     = (float*)d_out;
  float* present = out + (size_t)NB * SEQ * D_MODEL;

  unsigned short* ws16 = (unsigned short*)d_ws;
  const size_t WSZ = (size_t)D_MODEL * D_MODEL;
  const size_t HSZ = (size_t)NB * NH * SEQ * DHEAD;
  unsigned short* WqT = ws16;
  unsigned short* WkT = ws16 + WSZ;
  unsigned short* WvT = ws16 + 2 * WSZ;
  unsigned short* WoT = ws16 + 3 * WSZ;
  unsigned short* qhB = ws16 + 4 * WSZ;
  unsigned short* khB = qhB + HSZ;
  unsigned short* vTB = khB + HSZ;
  unsigned short* ctx = vTB + HSZ;

  int tgrid = (int)(WSZ / 256);
  transpose_w_bf16<<<tgrid, 256, 0, stream>>>(Wq, WqT);
  transpose_w_bf16<<<tgrid, 256, 0, stream>>>(Wk, WkT);
  transpose_w_bf16<<<tgrid, 256, 0, stream>>>(Wv, WvT);
  transpose_w_bf16<<<tgrid, 256, 0, stream>>>(Wo, WoT);

  dim3 blk(32, 4);
  qkv_proj<<<4096, blk, 0, stream>>>(q, WqT, bq, qhB, nullptr, 0);
  qkv_proj<<<4096, blk, 0, stream>>>(k, WkT, bk, khB, present, 0);
  qkv_proj<<<4096, blk, 0, stream>>>(v, WvT, bv, vTB, present, 1);

  flash_attn<<<1024, blk, 0, stream>>>(qhB, khB, vTB, ctx);

  out_proj<<<4096, blk, 0, stream>>>(ctx, WoT, bo, out);
}
